// SmallBitConvNet_57492432224587
// MI455X (gfx1250) — matrix operand order repaired and checked by reading, not compiled, not run on hardware
//
#include <hip/hip_runtime.h>

// ---------------- CDNA5 WMMA / TDM types ----------------
typedef __attribute__((ext_vector_type(16))) _Float16 v16h;
typedef __attribute__((ext_vector_type(8)))  _Float16 v8h;
typedef __attribute__((ext_vector_type(8)))  float    v8f;
typedef __attribute__((ext_vector_type(4)))  unsigned int u32x4;
typedef __attribute__((ext_vector_type(8)))  int      i32x8;
typedef __attribute__((ext_vector_type(4)))  int      i32x4;

union AFrag {
    v16h v;
    v8h  h[2];
    _Float16 e[16];
};

__device__ __forceinline__ v8f wmma_f16(const v16h& a, const v16h& b, const v8f& c) {
    // (neg_a, A, neg_b, B, c_mod, C, reuse_a, reuse_b)
    return __builtin_amdgcn_wmma_f32_16x16x32_f16(false, a, false, b, (short)0, c, false, false);
}

// Tensor Data Mover: stage a 2D tile (tile_w x tile_h elements, 2B each) from a
// row-major global matrix (row pitch = stride elements) into LDS at lds_off.
// D# layout per CDNA5 ISA (Group0 128b, Group1 256b).
// 6-arg builtin form (clang-23 / therock-10.0 toolchain).
__device__ __forceinline__ void tdm_load_2d(const void* gaddr, unsigned int lds_off,
                                            unsigned int tw, unsigned int th,
                                            unsigned int tile_w, unsigned int tile_h,
                                            unsigned int stride) {
    unsigned long long ga = (unsigned long long)(uintptr_t)gaddr;
    u32x4 g0;
    g0[0] = 1u;                                            // count=1, user descriptor
    g0[1] = lds_off;                                       // lds_addr (bytes)
    g0[2] = (unsigned int)ga;                              // global_addr[31:0]
    g0[3] = (unsigned int)((ga >> 32) & 0x01FFFFFFu) | (2u << 30);  // addr[56:32] | type=2
    i32x8 g1;
    g1[0] = 0x00010000;                                    // wg_mask=0, data_size=1 (2B)
    g1[1] = (int)(tw << 16);                               // tensor_dim0[15:0] in [31:16]
    g1[2] = (int)((tw >> 16) | (th << 16));                // dim0 hi16 | tensor_dim1 lo16
    g1[3] = (int)((th >> 16) | (tile_w << 16));            // dim1 hi16 | tile_dim0
    g1[4] = (int)(tile_h & 0xFFFFu);                       // tile_dim1 | tile_dim2=0
    g1[5] = (int)stride;                                   // tensor_dim0_stride[31:0]
    g1[6] = 0;                                             // stride hi | dim1_stride lo
    g1[7] = 0;
    i32x4 z4 = {0, 0, 0, 0};
    i32x8 z8 = {0, 0, 0, 0, 0, 0, 0, 0};
    __builtin_amdgcn_tensor_load_to_lds(g0, g1, z4, z4, z8, 0);
}

// ---------------- workspace layout (bytes) ----------------
static constexpr size_t OFF_ALPHA = 0;                                  // 4 f32
static constexpr size_t OFF_STATS = 256;                                // 896 f32 (s1:128, s2:256, s3:512)
static constexpr size_t OFF_SB    = OFF_STATS + 4096;                   // 896 f32 scale/bias
static constexpr size_t OFF_WQ1   = OFF_SB + 4096;                      // 64*32 f16
static constexpr size_t OFF_WQ2   = OFF_WQ1 + 4096;                     // 128*576 f16
static constexpr size_t OFF_WF1Q  = OFF_WQ2 + 147456;                   // 256*8192 f16
static constexpr size_t OFF_WF2Q  = OFF_WF1Q + 4194304;                 // 10*256 f32
static constexpr size_t OFF_H1    = OFF_WF2Q + 16384;                   // 1024*32*32*64 f16 (NHWC)
static constexpr size_t OFF_P1    = OFF_H1 + 134217728;                 // 1024*18*18*64 f16 (padded NHWC)
static constexpr size_t OFF_H2    = OFF_P1 + 42467328;                  // 1024*16*16*128 f16 (NHWC)
static constexpr size_t OFF_P2    = OFF_H2 + 67108864;                  // 1024*8*8*128 f16 (= FC1 A matrix)
static constexpr size_t OFF_H3    = OFF_P2 + 16777216;                  // 1024*256 f32

// ---------------- utility kernels ----------------
__global__ void zero_f32_kernel(float* __restrict__ p, int n) {
    int i = blockIdx.x * blockDim.x + threadIdx.x;
    if (i < n) p[i] = 0.f;
}

__global__ void zero_f16_kernel(_Float16* __restrict__ p, size_t n) {
    size_t i = (size_t)blockIdx.x * blockDim.x + threadIdx.x;
    if (i < n) p[i] = (_Float16)0.f;
}

__global__ __launch_bounds__(256) void absmean_kernel(const float* __restrict__ w, int n,
                                                      float* __restrict__ alpha) {
    __shared__ float red[256];
    float s = 0.f;
    for (int i = threadIdx.x; i < n; i += 256) s += fabsf(w[i]);
    red[threadIdx.x] = s;
    __syncthreads();
    for (int off = 128; off > 0; off >>= 1) {
        if ((int)threadIdx.x < off) red[threadIdx.x] += red[threadIdx.x + off];
        __syncthreads();
    }
    if (threadIdx.x == 0) *alpha = red[0] / (float)n;
}

__device__ __forceinline__ float signf(float t) { return t > 0.f ? 1.f : (t < 0.f ? -1.f : 0.f); }

// w1 [64][3][3][3] OIHW -> wq1 [64][32] f16, k = c*9 + dy*3 + dx, pad 27..31 = 0
__global__ void quant_w1_kernel(const float* __restrict__ w, const float* __restrict__ alphas,
                                _Float16* __restrict__ wq) {
    int i = blockIdx.x * blockDim.x + threadIdx.x;
    if (i >= 64 * 32) return;
    int o = i >> 5, k = i & 31;
    float a = alphas[0];
    float v = 0.f;
    if (k < 27) v = signf(w[o * 27 + k]) * a;
    wq[i] = (_Float16)v;
}

// w2 [128][64][3][3] OIHW -> wq2 [128][576] f16 with k = (dy*3+dx)*64 + c
__global__ void quant_w2_kernel(const float* __restrict__ w, const float* __restrict__ alphas,
                                _Float16* __restrict__ wq) {
    int i = blockIdx.x * blockDim.x + threadIdx.x;
    if (i >= 128 * 576) return;
    int o = i / 576, k = i % 576;
    int cell = k >> 6, c = k & 63;                 // cell = dy*3+dx
    float a = alphas[1];
    wq[i] = (_Float16)(signf(w[o * 576 + c * 9 + cell]) * a);
}

// wfc1 [256][8192] with k = c*64 + s (NCHW flatten) -> wq [256][8192] with k' = s*128 + c (NHWC)
__global__ void quant_fc1_kernel(const float* __restrict__ w, const float* __restrict__ alphas,
                                 _Float16* __restrict__ wq) {
    int i = blockIdx.x * blockDim.x + threadIdx.x;
    if (i >= 256 * 8192) return;
    int n = i >> 13, k = i & 8191;
    int s = k >> 7, c = k & 127;
    float a = alphas[2];
    wq[i] = (_Float16)(signf(w[(size_t)n * 8192 + c * 64 + s]) * a);
}

__global__ void quant_fc2_kernel(const float* __restrict__ w, const float* __restrict__ alphas,
                                 float* __restrict__ wq) {
    int i = blockIdx.x * blockDim.x + threadIdx.x;
    if (i >= 10 * 256) return;
    wq[i] = signf(w[i]) * alphas[3];
}

__global__ void bn_finalize_kernel(const float* __restrict__ stats, const float* __restrict__ g,
                                   const float* __restrict__ b, float invcnt,
                                   float* __restrict__ sc, float* __restrict__ bi, int C) {
    int c = threadIdx.x;
    if (c < C) {
        float mean = stats[c] * invcnt;
        float var  = stats[C + c] * invcnt - mean * mean;
        float s    = g[c] * rsqrtf(var + 1e-5f);
        sc[c] = s;
        bi[c] = b[c] - mean * s;
    }
}

// ---------------- conv1: implicit GEMM, M=B*1024, N=64, K=27(pad 32) ----------------
__global__ __launch_bounds__(32) void conv1_wmma_kernel(const float* __restrict__ xin,
                                                        const _Float16* __restrict__ wq1,
                                                        _Float16* __restrict__ h1,
                                                        float* __restrict__ stats1) {
    __shared__ _Float16 smA[16 * 32];
    const int lane = threadIdx.x;
    const int tile = blockIdx.x;            // 65536 tiles of 16 rows

    // im2col into LDS: entry (row=t, k=lane)
    for (int t = 0; t < 16; ++t) {
        int gr = tile * 16 + t;
        int b = gr >> 10, pos = gr & 1023;
        int y = pos >> 5, xx = pos & 31;
        int k = lane;
        float v = 0.f;
        if (k < 27) {
            int c = k / 9, r = k % 9, dy = r / 3, dx = r % 3;
            int gy = y + dy - 1, gx = xx + dx - 1;
            if (gy >= 0 && gy < 32 && gx >= 0 && gx < 32)
                v = xin[(((size_t)b * 3 + c) * 32 + gy) * 32 + gx];
        }
        smA[t * 32 + k] = (_Float16)v;
    }
    __syncthreads();

    AFrag a;
    {
        int m = lane & 15, kb = (lane >> 4) * 16;
        for (int i = 0; i < 8; ++i) {
            a.e[i]     = smA[m * 32 + kb + i];
            a.e[i + 8] = smA[m * 32 + kb + 8 + i];
        }
    }
    const int n0 = lane & 15, kb2 = (lane >> 4) * 16;
    v8f acc[4];
    for (int t = 0; t < 4; ++t) {
        v16h bf = *(const v16h*)(wq1 + (t * 16 + n0) * 32 + kb2);
        v8f c = {};
        acc[t] = wmma_f16(a.v, bf, c);
    }
    const int mhi = (lane >> 4) * 8;
    for (int t = 0; t < 4; ++t) {
        int n = t * 16 + n0;
        float s = 0.f, sq = 0.f;
        for (int j = 0; j < 8; ++j) {
            int gr = tile * 16 + mhi + j;
            float d = acc[t][j];
            h1[(size_t)gr * 64 + n] = (_Float16)d;
            s += d; sq += d * d;
        }
        atomicAdd(&stats1[n], s);
        atomicAdd(&stats1[64 + n], sq);
    }
}

// ---------------- BN1 + ReLU + maxpool -> padded NHWC f16 ----------------
__global__ __launch_bounds__(256) void pool1_kernel(const _Float16* __restrict__ h1,
                                                    const float* __restrict__ sc,
                                                    const float* __restrict__ bi,
                                                    _Float16* __restrict__ p1) {
    int idx = blockIdx.x * 256 + threadIdx.x;       // 1024*16*16*64, exact
    int c = idx & 63, x = (idx >> 6) & 15, y = (idx >> 10) & 15, b = idx >> 14;
    float s = sc[c], t = bi[c];
    float m = 0.f;                                  // relu output is >= 0
    for (int dy = 0; dy < 2; ++dy)
        for (int dx = 0; dx < 2; ++dx) {
            float v = (float)h1[(((size_t)b * 32 + 2 * y + dy) * 32 + 2 * x + dx) * 64 + c] * s + t;
            if (v > m) m = v;
        }
    p1[(((size_t)b * 18 + y + 1) * 18 + (x + 1)) * 64 + c] = (_Float16)m;
}

// ---------------- conv2: implicit GEMM, M=B*256, N=128, K=576 ----------------
// B (weights) staged by the Tensor Data Mover into a double-buffered LDS tile,
// synchronized with TENSORcnt; A fragments are direct 16B global loads.
__global__ __launch_bounds__(32) void conv2_wmma_kernel(const _Float16* __restrict__ p1,
                                                        const _Float16* __restrict__ wq2,
                                                        _Float16* __restrict__ h2,
                                                        float* __restrict__ stats2) {
    __shared__ _Float16 smB[2][128 * 32];           // 2 x 8KB double buffer
    const unsigned int lds0 = (unsigned int)(uintptr_t)&smB[0][0];
    const unsigned int lds1 = (unsigned int)(uintptr_t)&smB[1][0];

    const int lane = threadIdx.x;
    const int tile = blockIdx.x;                    // 16384 tiles of 16 rows
    const int m = lane & 15;
    const int gr = tile * 16 + m;
    const int b = gr >> 8, y = (gr >> 4) & 15, xx = gr & 15;
    const _Float16* base = p1 + (size_t)b * 18 * 18 * 64;
    const int off = (lane >> 4) * 8;
    const int kb2 = (lane >> 4) * 16;
    const int n0 = lane & 15;

    v8f acc[8];
    for (int t = 0; t < 8; ++t) acc[t] = (v8f){0.f, 0.f, 0.f, 0.f, 0.f, 0.f, 0.f, 0.f};

    // prologue: stage K-chunk 0 (all 128 rows x 32 cols of wq2)
    tdm_load_2d(wq2, lds0, 576u, 128u, 32u, 128u, 576u);

    for (int s = 0; s < 18; ++s) {
        if (s + 1 < 18) {
            tdm_load_2d(wq2 + (s + 1) * 32, ((s + 1) & 1) ? lds1 : lds0,
                        576u, 128u, 32u, 128u, 576u);
            __builtin_amdgcn_s_wait_tensorcnt((short)1);   // chunk s complete
        } else {
            __builtin_amdgcn_s_wait_tensorcnt((short)0);
        }
        AFrag a;
        for (int h = 0; h < 2; ++h) {
            int kk = s * 32 + kb2 + h * 8;
            int cell = kk >> 6, c0 = kk & 63;       // cell = dy*3+dx, c0 = channel base
            int dy = cell / 3, dx = cell % 3;
            const _Float16* ap = base + (((y + dy) * 18) + (xx + dx)) * 64 + c0;
            a.h[h] = *(const v8h*)ap;
        }
        const _Float16* bb = smB[s & 1];
#pragma unroll
        for (int t = 0; t < 8; ++t) {
            v16h bf = *(const v16h*)(bb + (t * 16 + n0) * 32 + kb2);
            acc[t] = wmma_f16(a.v, bf, acc[t]);
        }
    }
    const int mhi = (lane >> 4) * 8;
    for (int t = 0; t < 8; ++t) {
        int n = t * 16 + n0;
        float su = 0.f, sq = 0.f;
        for (int j = 0; j < 8; ++j) {
            int gr2 = tile * 16 + mhi + j;
            float d = acc[t][j];
            h2[(size_t)gr2 * 128 + n] = (_Float16)d;
            su += d; sq += d * d;
        }
        atomicAdd(&stats2[n], su);
        atomicAdd(&stats2[128 + n], sq);
    }
}

// ---------------- BN2 + ReLU + maxpool -> FC1 A-matrix [1024][8192] f16 ----------------
__global__ __launch_bounds__(256) void pool2_kernel(const _Float16* __restrict__ h2,
                                                    const float* __restrict__ sc,
                                                    const float* __restrict__ bi,
                                                    _Float16* __restrict__ p2) {
    int idx = blockIdx.x * 256 + threadIdx.x;       // 1024*8*8*128, exact
    int c = idx & 127, x = (idx >> 7) & 7, y = (idx >> 10) & 7, b = idx >> 13;
    float s = sc[c], t = bi[c];
    float m = 0.f;
    for (int dy = 0; dy < 2; ++dy)
        for (int dx = 0; dx < 2; ++dx) {
            float v = (float)h2[(((size_t)b * 16 + 2 * y + dy) * 16 + 2 * x + dx) * 128 + c] * s + t;
            if (v > m) m = v;
        }
    p2[(size_t)b * 8192 + (y * 8 + x) * 128 + c] = (_Float16)m;
}

// ---------------- FC1: M=1024, N=256 (64 per block), K=8192 ----------------
// B tile (64 rows x 32 cols) staged per K-step by TDM, double-buffered in LDS.
__global__ __launch_bounds__(32) void fc1_wmma_kernel(const _Float16* __restrict__ p2,
                                                      const _Float16* __restrict__ wq,
                                                      float* __restrict__ h3,
                                                      float* __restrict__ stats3) {
    __shared__ _Float16 smB[2][64 * 32];            // 2 x 4KB double buffer
    const unsigned int lds0 = (unsigned int)(uintptr_t)&smB[0][0];
    const unsigned int lds1 = (unsigned int)(uintptr_t)&smB[1][0];

    const int lane = threadIdx.x;
    const int mtile = blockIdx.x;                   // 64
    const int nbase = blockIdx.y * 64;              // 4
    const int m = lane & 15;
    const size_t arow = (size_t)(mtile * 16 + m) * 8192;
    const int kb = (lane >> 4) * 8, kb2 = (lane >> 4) * 16;
    const int n0 = lane & 15;
    const _Float16* wbase = wq + (size_t)nbase * 8192;

    v8f acc[4];
    for (int t = 0; t < 4; ++t) acc[t] = (v8f){0.f, 0.f, 0.f, 0.f, 0.f, 0.f, 0.f, 0.f};

    tdm_load_2d(wbase, lds0, 8192u, 256u, 32u, 64u, 8192u);

    for (int s = 0; s < 256; ++s) {
        if (s + 1 < 256) {
            tdm_load_2d(wbase + (s + 1) * 32, ((s + 1) & 1) ? lds1 : lds0,
                        8192u, 256u, 32u, 64u, 8192u);
            __builtin_amdgcn_s_wait_tensorcnt((short)1);
        } else {
            __builtin_amdgcn_s_wait_tensorcnt((short)0);
        }
        AFrag a;
        a.h[0] = *(const v8h*)(p2 + arow + s * 32 + kb2);
        a.h[1] = *(const v8h*)(p2 + arow + s * 32 + kb2 + 8);
        if (s + 8 < 256)
            __builtin_prefetch((const void*)(p2 + arow + (s + 8) * 32 + kb), 0, 1);
        const _Float16* bb = smB[s & 1];
#pragma unroll
        for (int t = 0; t < 4; ++t) {
            v16h bf = *(const v16h*)(bb + (t * 16 + n0) * 32 + kb2);
            acc[t] = wmma_f16(a.v, bf, acc[t]);
        }
    }
    const int mhi = (lane >> 4) * 8;
    for (int t = 0; t < 4; ++t) {
        int n = nbase + t * 16 + n0;
        float su = 0.f, sq = 0.f;
        for (int j = 0; j < 8; ++j) {
            int gr2 = mtile * 16 + mhi + j;
            float d = acc[t][j];
            h3[(size_t)gr2 * 256 + n] = d;
            su += d; sq += d * d;
        }
        atomicAdd(&stats3[n], su);
        atomicAdd(&stats3[256 + n], sq);
    }
}

// ---------------- BN3 + ReLU + FC2 (256 -> 10) ----------------
__global__ __launch_bounds__(32) void fc2_kernel(const float* __restrict__ h3,
                                                 const float* __restrict__ sc,
                                                 const float* __restrict__ bi,
                                                 const float* __restrict__ wq,
                                                 float* __restrict__ out) {
    int b = blockIdx.x, lane = threadIdx.x;
    float acc[10];
    for (int j = 0; j < 10; ++j) acc[j] = 0.f;
    for (int c = lane; c < 256; c += 32) {
        float v = h3[(size_t)b * 256 + c] * sc[c] + bi[c];
        v = v > 0.f ? v : 0.f;
        for (int j = 0; j < 10; ++j) acc[j] += v * wq[j * 256 + c];
    }
    for (int off = 16; off > 0; off >>= 1)
        for (int j = 0; j < 10; ++j) acc[j] += __shfl_down(acc[j], off, 32);
    if (lane == 0)
        for (int j = 0; j < 10; ++j) out[b * 10 + j] = acc[j];
}

// ---------------- launch ----------------
extern "C" void kernel_launch(void* const* d_in, const int* in_sizes, int n_in,
                              void* d_out, int out_size, void* d_ws, size_t ws_size,
                              hipStream_t stream) {
    (void)in_sizes; (void)n_in; (void)out_size; (void)ws_size;
    const float* x    = (const float*)d_in[0];
    const float* w1   = (const float*)d_in[1];
    const float* g1   = (const float*)d_in[2];
    const float* b1   = (const float*)d_in[3];
    const float* w2   = (const float*)d_in[4];
    const float* g2   = (const float*)d_in[5];
    const float* b2   = (const float*)d_in[6];
    const float* wfc1 = (const float*)d_in[7];
    const float* g3   = (const float*)d_in[8];
    const float* b3   = (const float*)d_in[9];
    const float* wfc2 = (const float*)d_in[10];
    float* out = (float*)d_out;
    char* ws = (char*)d_ws;

    float*     alphas = (float*)(ws + OFF_ALPHA);
    float*     stats  = (float*)(ws + OFF_STATS);
    float*     st1 = stats, *st2 = stats + 128, *st3 = stats + 384;
    float*     sb  = (float*)(ws + OFF_SB);
    float      *sc1 = sb, *bi1 = sb + 64, *sc2 = sb + 128, *bi2 = sb + 256, *sc3 = sb + 384, *bi3 = sb + 640;
    _Float16*  wq1   = (_Float16*)(ws + OFF_WQ1);
    _Float16*  wq2   = (_Float16*)(ws + OFF_WQ2);
    _Float16*  wf1q  = (_Float16*)(ws + OFF_WF1Q);
    float*     wf2q  = (float*)(ws + OFF_WF2Q);
    _Float16*  h1    = (_Float16*)(ws + OFF_H1);
    _Float16*  p1    = (_Float16*)(ws + OFF_P1);
    _Float16*  h2    = (_Float16*)(ws + OFF_H2);
    _Float16*  p2    = (_Float16*)(ws + OFF_P2);
    float*     h3    = (float*)(ws + OFF_H3);

    // init: zero BN stats + padded p1 buffer (border must be zero)
    zero_f32_kernel<<<(896 + 255) / 256, 256, 0, stream>>>(stats, 896);
    const size_t p1n = (size_t)1024 * 18 * 18 * 64;
    zero_f16_kernel<<<(unsigned)((p1n + 255) / 256), 256, 0, stream>>>(p1, p1n);

    // alphas = mean(|w|)
    absmean_kernel<<<1, 256, 0, stream>>>(w1, 64 * 27, alphas + 0);
    absmean_kernel<<<1, 256, 0, stream>>>(w2, 128 * 576, alphas + 1);
    absmean_kernel<<<1, 256, 0, stream>>>(wfc1, 256 * 8192, alphas + 2);
    absmean_kernel<<<1, 256, 0, stream>>>(wfc2, 10 * 256, alphas + 3);

    // quantize + pack weights into WMMA B layouts
    quant_w1_kernel <<<(64 * 32 + 255) / 256, 256, 0, stream>>>(w1, alphas, wq1);
    quant_w2_kernel <<<(128 * 576 + 255) / 256, 256, 0, stream>>>(w2, alphas, wq2);
    quant_fc1_kernel<<<(256 * 8192 + 255) / 256, 256, 0, stream>>>(wfc1, alphas, wf1q);
    quant_fc2_kernel<<<(10 * 256 + 255) / 256, 256, 0, stream>>>(wfc2, alphas, wf2q);

    // layer 1
    conv1_wmma_kernel<<<65536, 32, 0, stream>>>(x, wq1, h1, st1);
    bn_finalize_kernel<<<1, 64, 0, stream>>>(st1, g1, b1, 1.f / 1048576.f, sc1, bi1, 64);
    pool1_kernel<<<65536, 256, 0, stream>>>(h1, sc1, bi1, p1);

    // layer 2
    conv2_wmma_kernel<<<16384, 32, 0, stream>>>(p1, wq2, h2, st2);
    bn_finalize_kernel<<<1, 128, 0, stream>>>(st2, g2, b2, 1.f / 262144.f, sc2, bi2, 128);
    pool2_kernel<<<32768, 256, 0, stream>>>(h2, sc2, bi2, p2);

    // FC head
    fc1_wmma_kernel<<<dim3(64, 4), 32, 0, stream>>>(p2, wf1q, h3, st3);
    bn_finalize_kernel<<<1, 256, 0, stream>>>(st3, g3, b3, 1.f / 1024.f, sc3, bi3, 256);
    fc2_kernel<<<1024, 32, 0, stream>>>(h3, sc3, bi3, wf2q, out);
}
